// ModulatedConv2d_532575945156
// MI455X (gfx1250) — matrix operand order repaired and checked by reading, not compiled, not run on hardware
//
#include <hip/hip_runtime.h>
#include <hip/hip_bf16.h>

typedef __attribute__((ext_vector_type(16))) __bf16        v16bf;
typedef __attribute__((ext_vector_type(8)))  float         v8f;
typedef __attribute__((ext_vector_type(4)))  unsigned int  v4u;   // also TDM group0
typedef __attribute__((ext_vector_type(8)))  int           i32x8; // TDM group1 / arg5
typedef __attribute__((ext_vector_type(4)))  int           i32x4; // TDM group2/3

#define N_BATCH   16
#define IN_CH     256
#define OUT_CH    256
#define HDIM      64
#define WDIM      64
#define STYLE_DIM 512
#define KTAPS     9
#define KDIM      (IN_CH * KTAPS)   /* 2304, reordered as tap*256+ic */
#define HW        (HDIM * WDIM)     /* 4096 */
#define PH        66                /* padded height */
#define PW        66                /* padded width */
#define PHW       (PH * PW)         /* 4356 */

#define BM   128
#define BN   128
#define BK   32
#define KITER (KDIM / BK)           /* 72 */
#define LDSK 40   /* padded LDS row stride: 64B row + 16B pad -> conflict-free b128 frags */

union FragBF  { v16bf bf; v4u u[2]; };
union FragOut { v8f f; float a[8]; };

// ---------------- Kernel 0: zero-pad + f32->bf16 convert x ----------------
__global__ __launch_bounds__(256)
void pad_convert_kernel(const float* __restrict__ x, __bf16* __restrict__ xpad) {
    const int plane = blockIdx.x;                 // n*IN_CH + ic
    const float* src = x + (size_t)plane * HW;
    __bf16* dst = xpad + (size_t)plane * PHW;
    for (int j = threadIdx.x; j < PHW; j += 256) {
        const int ph = j / PW;
        const int pw = j - ph * PW;
        float v = 0.f;
        if (ph >= 1 && ph <= HDIM && pw >= 1 && pw <= WDIM)
            v = src[(ph - 1) * WDIM + (pw - 1)];
        dst[j] = (__bf16)v;
    }
}

// ---------------- Kernel 1: s[n][ic] = style[n] . mod_w[ic] + mod_b[ic] ----------------
__global__ __launch_bounds__(IN_CH)
void mod_linear_kernel(const float* __restrict__ style,
                       const float* __restrict__ mod_w,
                       const float* __restrict__ mod_b,
                       float* __restrict__ s_out) {
    __shared__ float styleS[STYLE_DIM];
    const int n = blockIdx.x;
    for (int i = threadIdx.x; i < STYLE_DIM; i += blockDim.x)
        styleS[i] = style[n * STYLE_DIM + i];
    __syncthreads();
    const int ic = threadIdx.x;
    const float* wrow = mod_w + ic * STYLE_DIM;
    float acc = 0.f;
    for (int d = 0; d < STYLE_DIM; ++d) acc += styleS[d] * wrow[d];
    s_out[n * IN_CH + ic] = acc + mod_b[ic];
}

// ---- Kernel 2: demodulate, emit bf16 A-matrix in TAP-MAJOR K order: w[n][oc][tap*256+ic] ----
__global__ __launch_bounds__(256)
void modulate_weight_kernel(const float* __restrict__ weight,
                            const float* __restrict__ s_in,
                            __bf16* __restrict__ w_bf16) {
    __shared__ float s1p[IN_CH];
    __shared__ float red[256];
    const int n  = blockIdx.y;
    const int oc = blockIdx.x;
    const int t  = threadIdx.x;

    s1p[t] = 1.0f + s_in[n * IN_CH + t];
    __syncthreads();

    const float* wrow = weight + (size_t)oc * KDIM;   // native layout: ic*9 + kh*3 + kw
    float ssq = 0.f;
    #pragma unroll
    for (int j = t; j < KDIM; j += 256) {
        float v = wrow[j] * s1p[j / KTAPS];
        ssq += v * v;
    }
    red[t] = ssq;
    __syncthreads();
    for (int off = 128; off > 0; off >>= 1) {
        if (t < off) red[t] += red[t + off];
        __syncthreads();
    }
    const float demod = rsqrtf(red[0] + 1e-8f);

    __bf16* orow = w_bf16 + (size_t)(n * OUT_CH + oc) * KDIM;
    #pragma unroll
    for (int j = t; j < KDIM; j += 256) {
        const int ic  = j / KTAPS;
        const int tap = j - ic * KTAPS;
        orow[tap * IN_CH + ic] = (__bf16)(wrow[j] * s1p[ic] * demod);
    }
}

// ---------------- Kernel 3: implicit-GEMM conv: TDM->LDS->WMMA, double-buffered ----------------
__global__ __launch_bounds__(256)
void modconv_gemm_kernel(const __bf16* __restrict__ x_pad,
                         const __bf16* __restrict__ w_bf16,
                         float* __restrict__ out) {
    __shared__ __bf16 As[2][BM][LDSK];   // [oc][k] tile, TDM-written with LDS padding
    __shared__ __bf16 Bs[2][BN][LDSK];   // [hw][k] tile (im2col from padded bf16 x)

    const int n      = blockIdx.z;
    const int ocBase = blockIdx.y * BM;
    const int hwBase = blockIdx.x * BN;
    const int h0     = hwBase >> 6;      // output row base of this hw tile (2 rows)

    const int t     = threadIdx.x;
    const int lane  = t & 31;
    const int wave  = t >> 5;        // 0..7
    const int waveM = wave & 3;
    const int waveN = wave >> 2;
    const int lo    = lane & 15;
    const int hi    = lane >> 4;

    v8f acc[2][4];
    const v8f vzero = {0.f, 0.f, 0.f, 0.f, 0.f, 0.f, 0.f, 0.f};
    #pragma unroll
    for (int mi = 0; mi < 2; ++mi)
        #pragma unroll
        for (int ni = 0; ni < 4; ++ni)
            acc[mi][ni] = vzero;

    // -------- A tile via Tensor Data Mover (one instruction per K-step, wave 0) --------
    const __bf16* aBase = w_bf16 + (size_t)(n * OUT_CH + ocBase) * KDIM;
    auto tdmStageA = [&](int kIdx, int p) {
        const unsigned long long ga = (unsigned long long)(aBase + (size_t)kIdx * BK);
        v4u g0;
        g0.x = 1u;                                       // count=1, user descriptor
        g0.y = (unsigned)(size_t)&As[p][0][0];           // lds_addr
        g0.z = (unsigned)(ga & 0xFFFFFFFFu);             // global_addr[31:0]
        g0.w = (unsigned)((ga >> 32) & 0x01FFFFFFu) | (2u << 30);  // addr[56:32] | type=2
        i32x8 g1;
        g1[0] = (1 << 16)      /* data_size = 2 bytes */
              | (1 << 20)      /* pad_enable */
              | (3 << 22)      /* pad_interval: 16 DWORDs (64B row) */
              | (3 << 25);     /* pad_amount:   4 DWORDs (16B)      */
        g1[1] = (int)(KDIM << 16);                 // tensor_dim0 lo16 (atomic addr 0)
        g1[2] = (int)(OUT_CH << 16);               // tensor_dim0 hi16=0 | tensor_dim1 lo16
        g1[3] = (int)(BK << 16);                   // tensor_dim1 hi16=0 | tile_dim0=32
        g1[4] = BM;                                // tile_dim1=128 | tile_dim2=0
        g1[5] = KDIM;                              // tensor_dim0_stride lo32
        g1[6] = 0;                                 // stride hi | dim1_stride (unused, 2D)
        g1[7] = 0;
        const i32x4 gz  = {0, 0, 0, 0};
        const i32x8 gz8 = {0, 0, 0, 0, 0, 0, 0, 0};
        __builtin_amdgcn_tensor_load_to_lds(g0, g1, gz, gz, gz8, 0);
    };

    // -------- B tile staging: straight bf16 copies from padded x (no div, no predication) ----
    const int kk0  = (t >> 4) * 2;          // even k-pair within tile (0..30)
    const int bNN0 = (t & 15) * 8;          // 8 consecutive hw per thread (stays in one row)
    const int brow = bNN0 >> 6;
    const int bw0  = bNN0 & 63;
    const unsigned short* xpn =
        (const unsigned short*)x_pad + (size_t)n * IN_CH * PHW;
    const size_t thrOff = (size_t)kk0 * PHW + (size_t)(h0 + brow) * PW + bw0;

    auto stageB = [&](int kIdx, int p) {
        const int tap = kIdx >> 3;           // block-uniform
        const int icb = kIdx & 7;
        const int kh  = tap / 3;             // uniform -> scalar
        const int kw  = tap - kh * 3;
        const unsigned short* src0 = xpn + thrOff + (size_t)(icb * 32) * PHW + kh * PW + kw;
        const unsigned short* src1 = src0 + PHW;   // ic+1
        #pragma unroll
        for (int j = 0; j < 8; ++j) {
            const unsigned lo16 = src0[j];
            const unsigned hi16 = src1[j];
            *(unsigned*)&Bs[p][bNN0 + j][kk0] = lo16 | (hi16 << 16);
        }
    };

    auto compute = [&](int p) {
        #pragma unroll
        for (int mi = 0; mi < 2; ++mi) {
            FragBF a;
            const int row = waveM * 32 + mi * 16 + lo;
            // A 16x32 bf16 (ISA 7.12.2): lanes 0-15: K0-7/K16-23; lanes 16-31: K8-15/K24-31
            a.u[0] = *(const v4u*)&As[p][row][hi * 8];
            a.u[1] = *(const v4u*)&As[p][row][16 + hi * 8];
            #pragma unroll
            for (int ni = 0; ni < 4; ++ni) {
                FragBF b;
                const int col = waveN * 64 + ni * 16 + lo;
                b.u[0] = *(const v4u*)&Bs[p][col][hi * 8];
                b.u[1] = *(const v4u*)&Bs[p][col][16 + hi * 8];
                acc[mi][ni] = __builtin_amdgcn_wmma_f32_16x16x32_bf16(
                    false, a.bf, false, b.bf, (short)0, acc[mi][ni], false, false);
            }
        }
    };

    // -------- double-buffered pipeline --------
    if (wave == 0) tdmStageA(0, 0);
    stageB(0, 0);
    __builtin_amdgcn_s_wait_tensorcnt((short)0);   // scalar; free for waves 1-7
    __syncthreads();

    for (int kIdx = 0; kIdx < KITER - 1; ++kIdx) {
        const int cur = kIdx & 1;
        const int nxt = cur ^ 1;
        if (wave == 0) tdmStageA(kIdx + 1, nxt);   // TDM overlaps WMMAs below
        stageB(kIdx + 1, nxt);
        compute(cur);
        __builtin_amdgcn_s_wait_tensorcnt((short)0);
        __syncthreads();
    }
    compute((KITER - 1) & 1);                      // last tile: no staging, no barrier

    // -------- epilogue: C frag layout: VGPR r -> M = r + 8*hi, N = lo --------
    float* outn = out + (size_t)(n * OUT_CH + ocBase) * HW + hwBase;
    #pragma unroll
    for (int mi = 0; mi < 2; ++mi) {
        #pragma unroll
        for (int ni = 0; ni < 4; ++ni) {
            FragOut fo; fo.f = acc[mi][ni];
            const int col = waveN * 64 + ni * 16 + lo;
            #pragma unroll
            for (int r = 0; r < 8; ++r) {
                const int row = waveM * 32 + mi * 16 + hi * 8 + r;
                outn[(size_t)row * HW + col] = fo.a[r];
            }
        }
    }
}

// ---------------------------------------------------------------------------
extern "C" void kernel_launch(void* const* d_in, const int* in_sizes, int n_in,
                              void* d_out, int out_size, void* d_ws, size_t ws_size,
                              hipStream_t stream) {
    const float* x      = (const float*)d_in[0];
    const float* style  = (const float*)d_in[1];
    const float* weight = (const float*)d_in[2];
    const float* mod_w  = (const float*)d_in[3];
    const float* mod_b  = (const float*)d_in[4];
    float* out = (float*)d_out;

    // Workspace: [ s: 16KB | w_bf16: 18.87MB | x_pad bf16: 35.68MB ]  (~54.6MB)
    float*  s_buf  = (float*)d_ws;
    __bf16* w_rord = (__bf16*)((char*)d_ws + 16384);
    __bf16* x_pad  = (__bf16*)((char*)d_ws + 16384 + (size_t)N_BATCH * OUT_CH * KDIM * 2);

    pad_convert_kernel<<<dim3(N_BATCH * IN_CH), dim3(256), 0, stream>>>(x, x_pad);
    mod_linear_kernel<<<dim3(N_BATCH), dim3(IN_CH), 0, stream>>>(style, mod_w, mod_b, s_buf);
    modulate_weight_kernel<<<dim3(OUT_CH, N_BATCH), dim3(256), 0, stream>>>(weight, s_buf, w_rord);
    modconv_gemm_kernel<<<dim3(HW / BN, OUT_CH / BM, N_BATCH), dim3(256), 0, stream>>>(x_pad, w_rord, out);
}
